// GraspNet_MSCG_context_seed_global_high_85787676770357
// MI455X (gfx1250) — matrix operand order repaired and checked by reading, not compiled, not run on hardware
//
#include <hip/hip_runtime.h>
#include <hip/hip_bf16.h>
#include <math.h>

#define B 2
#define NPTS 15000
#define MPT 1024
#define CSEED 512
#define K1DIM 515
#define K1PAD 544   // 17 * 32, zero-padded K for GEMM1

typedef __attribute__((ext_vector_type(16))) _Float16 v16h;
typedef __attribute__((ext_vector_type(8)))  float    v8f;

union FragU { v16h v; uint4 q[2]; };

// A fragment (16xK tile of row-major [M][K] f16, ldk halves).
// ISA layout (16-bit A 16x32): lanes 0-15: M=lane, K = {0..7, 16..23};
// lanes 16-31: M=lane-16, K = {8..15, 24..31}. Element e: K = (e<8?e:8+e)+8*hi.
__device__ inline v16h load_frag_a(const _Float16* p, int ldk) {
  int lane = threadIdx.x & 31;
  const _Float16* q = p + (size_t)(lane & 15) * ldk + (lane >> 4) * 8;
  FragU u;
  u.q[0] = *(const uint4*)(q);        // K base+0..7
  u.q[1] = *(const uint4*)(q + 16);   // K base+16..23
  return u.v;
}
// B fragment (Kx16 tile). Storage is column-major: [N][K] with stride ldk.
// ISA layout (16-bit B 32x16): lanes 0-15: N=lane, K = {0..7, 16..23};
// lanes 16-31: N=lane-16, K = {8..15, 24..31}.
__device__ inline v16h load_frag_b(const _Float16* p, int ldk) {
  int lane = threadIdx.x & 31;
  const _Float16* q = p + (size_t)(lane & 15) * ldk + (lane >> 4) * 8;
  FragU u;
  u.q[0] = *(const uint4*)(q);
  u.q[1] = *(const uint4*)(q + 16);
  return u.v;
}

// ---------------- masked farthest point sampling -----------------
__global__ __launch_bounds__(1024)
void fps_kernel(const float* __restrict__ xyz, const int* __restrict__ mask,
                int* __restrict__ fps_idx) {
  int b = blockIdx.x;
  int tid = threadIdx.x, lane = tid & 31, wid = tid >> 5;
  const float* X = xyz + (size_t)b * NPTS * 3;
  const int*   Mk = mask + (size_t)b * NPTS;
  __shared__ float swv[32];
  __shared__ int   swi[32];
  __shared__ float sfx, sfy, sfz;
  __shared__ int   sfar;

  float d[15], px[15], py[15], pz[15];
  bool  mm[15];
  int cand = 0x7fffffff;
#pragma unroll
  for (int s = 0; s < 15; ++s) {
    int p = tid + s * 1024;
    bool inb = p < NPTS;
    int pp = inb ? p : 0;
    bool m = inb && (Mk[pp] != 0);
    px[s] = X[pp * 3 + 0]; py[s] = X[pp * 3 + 1]; pz[s] = X[pp * 3 + 2];
    mm[s] = m;
    d[s] = m ? 1e10f : (inb ? -1.f : -2.f);
    if (m && p < cand) cand = p;
  }
  // start = argmax(mask) == first masked index (0 if none)
  for (int off = 16; off; off >>= 1) { int o = __shfl_down(cand, off); cand = min(cand, o); }
  if (lane == 0) swi[wid] = cand;
  __syncthreads();
  if (wid == 0) {
    int c = swi[lane];
    for (int off = 16; off; off >>= 1) { int o = __shfl_down(c, off); c = min(c, o); }
    if (lane == 0) sfar = (c == 0x7fffffff) ? 0 : c;
  }
  __syncthreads();
  int far = sfar;

  for (int it = 0; it < MPT; ++it) {
    if (tid == 0) {
      fps_idx[b * MPT + it] = far;
      sfx = X[far * 3 + 0]; sfy = X[far * 3 + 1]; sfz = X[far * 3 + 2];
    }
    __syncthreads();
    float fx = sfx, fy = sfy, fz = sfz;
    float bv = -3.f; int bi = 0x7fffffff;
#pragma unroll
    for (int s = 0; s < 15; ++s) {
      float dx = px[s] - fx, dy = py[s] - fy, dz = pz[s] - fz;
      float dd = dx * dx + dy * dy + dz * dz;
      if (mm[s]) d[s] = fminf(d[s], dd);
      int p = tid + s * 1024;
      if (d[s] > bv || (d[s] == bv && p < bi)) { bv = d[s]; bi = p; }
    }
    for (int off = 16; off; off >>= 1) {
      float ov = __shfl_down(bv, off); int oi = __shfl_down(bi, off);
      if (ov > bv || (ov == bv && oi < bi)) { bv = ov; bi = oi; }
    }
    if (lane == 0) { swv[wid] = bv; swi[wid] = bi; }
    __syncthreads();
    if (wid == 0) {
      float v2 = swv[lane]; int i2 = swi[lane];
      for (int off = 16; off; off >>= 1) {
        float ov = __shfl_down(v2, off); int oi = __shfl_down(i2, off);
        if (ov > v2 || (ov == v2 && oi < i2)) { v2 = ov; i2 = oi; }
      }
      if (lane == 0) sfar = i2;
    }
    __syncthreads();
    far = sfar;
  }
}

// ------------- gather sampled xyz (f32) + features (f16, j-major) -------------
__global__ void gather_kernel(const float* __restrict__ seed_xyz,
                              const float* __restrict__ seed_feat,
                              const int* __restrict__ fps_idx,
                              float* __restrict__ xyz_g,
                              _Float16* __restrict__ featT) {
  int j = blockIdx.x, b = blockIdx.y, c = threadIdx.x;   // 512 threads
  int idx = fps_idx[b * MPT + j];
  featT[((size_t)(b * MPT + j)) * CSEED + c] =
      (_Float16)seed_feat[((size_t)b * CSEED + c) * NPTS + idx];
  if (c < 3)
    xyz_g[(b * MPT + j) * 3 + c] = seed_xyz[((size_t)b * NPTS + idx) * 3 + c];
}

// ---------------- weight conversions ----------------
__global__ void convert_w1_kernel(const float* __restrict__ src, _Float16* __restrict__ dst) {
  int i = blockIdx.x * 256 + threadIdx.x;     // over 4*256*544
  if (i >= 4 * 256 * K1PAD) return;
  int col = i % K1PAD, row = i / K1PAD;
  dst[i] = (col < K1DIM) ? (_Float16)src[(size_t)row * K1DIM + col] : (_Float16)0.f;
}
__global__ void convert_f16_kernel(const float* __restrict__ src, _Float16* __restrict__ dst, int n) {
  int i = blockIdx.x * 256 + threadIdx.x;
  if (i < n) dst[i] = (_Float16)src[i];
}

// -------- local coords + cylinder query + stable first-ns selection --------
__global__ __launch_bounds__(1024)
void local_group_kernel(const float* __restrict__ xyz_g,
                        const float* __restrict__ views_rot,
                        int* __restrict__ grp_base,
                        _Float16* __restrict__ gxyz_base) {
  int j = blockIdx.x, b = blockIdx.y;
  int tid = threadIdx.x, lane = tid & 31, wid = tid >> 5;
  __shared__ float sR[9], sPj[3];
  __shared__ int scnt[32], soff[32], stot;
  __shared__ int sfk; __shared__ float sf[3];
  __shared__ float s0[3];
  if (tid < 9) sR[tid] = views_rot[((size_t)(b * MPT + j)) * 9 + tid];
  if (tid < 3) sPj[tid] = xyz_g[(b * MPT + j) * 3 + tid];
  __syncthreads();
  int k = tid;
  float rx = xyz_g[(b * MPT + k) * 3 + 0] - sPj[0];
  float ry = xyz_g[(b * MPT + k) * 3 + 1] - sPj[1];
  float rz = xyz_g[(b * MPT + k) * 3 + 2] - sPj[2];
  float l0 = rx * sR[0] + ry * sR[3] + rz * sR[6];
  float l1 = rx * sR[1] + ry * sR[4] + rz * sR[7];
  float l2 = rx * sR[2] + ry * sR[5] + rz * sR[8];
  float r2 = l1 * l1 + l2 * l2;
  bool in_h = (l0 >= -0.02f) && (l0 <= 0.06f);
  if (tid == 0) { s0[0] = l0; s0[1] = l1; s0[2] = l2; }

  const float rad2v[4] = {0.025f * 0.025f, 0.0375f * 0.0375f, 0.05f * 0.05f, 0.075f * 0.075f};
  const int nsv[4]  = {16, 32, 32, 64};
  const int roff[4] = {0, 2048 * 16, 2048 * 48, 2048 * 80};  // sample-slot offsets

  for (int s = 0; s < 4; ++s) {
    int ns = nsv[s];
    bool valid = in_h && (r2 < rad2v[s]);
    unsigned bal = (unsigned)__ballot(valid);
    int wrank = __popc(bal & ((1u << lane) - 1u));
    if (lane == 0) scnt[wid] = __popc(bal);
    __syncthreads();
    if (wid == 0) {
      int c = scnt[lane];
      int inc = c;
      for (int off = 1; off < 32; off <<= 1) { int o = __shfl_up(inc, off); if (lane >= off) inc += o; }
      soff[lane] = inc - c;
      if (lane == 31) stot = inc;
    }
    __syncthreads();
    int rank = soff[wid] + wrank;
    int cnt = stot;
    int* gp = grp_base + roff[s] + (size_t)(b * MPT + j) * ns;
    _Float16* gx = gxyz_base + ((size_t)roff[s] + (size_t)(b * MPT + j) * ns) * 3;
    if (valid && rank == 0) { sfk = k; sf[0] = l0; sf[1] = l1; sf[2] = l2; }
    __syncthreads();
    if (valid && rank < ns) {
      gp[rank] = k;
      gx[rank * 3 + 0] = (_Float16)l0; gx[rank * 3 + 1] = (_Float16)l1; gx[rank * 3 + 2] = (_Float16)l2;
    }
    if (tid >= cnt && tid < ns) {
      int fk; float f0, f1, f2;
      if (cnt > 0) { fk = sfk; f0 = sf[0]; f1 = sf[1]; f2 = sf[2]; }
      else         { fk = 0;   f0 = s0[0]; f1 = s0[1]; f2 = s0[2]; }
      gp[tid] = fk;
      gx[tid * 3 + 0] = (_Float16)f0; gx[tid * 3 + 1] = (_Float16)f1; gx[tid * 3 + 2] = (_Float16)f2;
    }
    __syncthreads();
  }
}

// -------- fused per-group MLP (GEMM1 -> ReLU -> GEMM2 -> ReLU -> max) --------
template <int NS>
__global__ __launch_bounds__(256)
void crop_mlp_kernel(const _Float16* __restrict__ W1f, const float* __restrict__ b1,
                     const _Float16* __restrict__ W2f, const float* __restrict__ b2,
                     const _Float16* __restrict__ featT,
                     const _Float16* __restrict__ gxyz,
                     const int* __restrict__ grp,
                     _Float16* __restrict__ vp, int ch_off) {
  constexpr int NT = NS / 16;
  __shared__ _Float16 Xl[NS][K1PAD];   // per-sample 544-vec, contiguous (col-major B)
  __shared__ _Float16 H1[NS][256];     // layer-1 output, col-major for GEMM2 B-frags
  int g = blockIdx.x;                  // b*1024 + j
  int tid = threadIdx.x, lane = tid & 31, w = tid >> 5;
  int base = (g >> 10) << 10;          // b*1024
  const int* gi = grp + (size_t)g * NS;

  for (int t = w; t < NS; t += 8) {
    int kidx = gi[t];
    const _Float16* src = featT + (size_t)(base + kidx) * CSEED;
    for (int c = lane; c < CSEED; c += 32) Xl[t][3 + c] = src[c];
    if (lane < 3) Xl[t][lane] = gxyz[((size_t)g * NS + t) * 3 + lane];
    for (int c = K1DIM + lane; c < K1PAD; c += 32) Xl[t][c] = (_Float16)0.f;
  }
  __syncthreads();

  // GEMM1: 256 x 544 x NS
  for (int tile = w; tile < 16 * NT; tile += 8) {
    int rt = tile / NT, ct = tile % NT;
    v8f acc = {};
    const _Float16* wrow = W1f + (size_t)rt * 16 * K1PAD;
    for (int kc = 0; kc < K1PAD / 32; ++kc) {
      v16h a  = load_frag_a(wrow + kc * 32, K1PAD);
      v16h bm = load_frag_b(&Xl[ct * 16][0] + kc * 32, K1PAD);
      acc = __builtin_amdgcn_wmma_f32_16x16x32_f16(false, a, false, bm, (short)0, acc, false, false);
    }
    int n = ct * 16 + (lane & 15);
    int mh = 8 * (lane >> 4);
#pragma unroll
    for (int r = 0; r < 8; ++r) {
      int M = rt * 16 + r + mh;
      float v = acc[r] + b1[M];
      v = v > 0.f ? v : 0.f;
      H1[n][M] = (_Float16)v;
    }
  }
  __syncthreads();

  // GEMM2: 512 x 256 x NS, then ReLU and max over the NS sample columns
  for (int rti = 0; rti < 4; ++rti) {
    int rt = w + rti * 8;
    float rowmax[8];
#pragma unroll
    for (int r = 0; r < 8; ++r) rowmax[r] = 0.f;   // ReLU outputs are >= 0
    const _Float16* wrow = W2f + (size_t)rt * 16 * 256;
    int mh = 8 * (lane >> 4);
    for (int ct = 0; ct < NT; ++ct) {
      v8f acc = {};
      for (int kc = 0; kc < 8; ++kc) {
        v16h a  = load_frag_a(wrow + kc * 32, 256);
        v16h bm = load_frag_b(&H1[ct * 16][0] + kc * 32, 256);
        acc = __builtin_amdgcn_wmma_f32_16x16x32_f16(false, a, false, bm, (short)0, acc, false, false);
      }
#pragma unroll
      for (int r = 0; r < 8; ++r) {
        float v = acc[r] + b2[rt * 16 + r + mh];
        v = v > 0.f ? v : 0.f;
        rowmax[r] = fmaxf(rowmax[r], v);
      }
    }
#pragma unroll
    for (int off = 1; off < 16; off <<= 1) {
#pragma unroll
      for (int r = 0; r < 8; ++r) rowmax[r] = fmaxf(rowmax[r], __shfl_xor(rowmax[r], off));
    }
    if ((lane & 15) == 0) {
#pragma unroll
      for (int r = 0; r < 8; ++r) {
        int M = rt * 16 + r + mh;
        vp[(size_t)g * 2048 + ch_off + M] = (_Float16)rowmax[r];
      }
    }
  }
}

// -------- fuse GEMM (K=2048) + gate GEMM (K=512) + sigmoid residual --------
__global__ __launch_bounds__(256)
void fuse_gate_kernel(const _Float16* __restrict__ fuseF, const float* __restrict__ fuse_b,
                      const _Float16* __restrict__ gateF, const float* __restrict__ gate_b,
                      const _Float16* __restrict__ vp, const _Float16* __restrict__ featT,
                      const float* __restrict__ seed_feat, const int* __restrict__ fps_idx,
                      float* __restrict__ out) {
  int tid = threadIdx.x, lane = tid & 31, w = tid >> 5;
  int rt = blockIdx.x;                        // 0..31 row-tiles of 512
  int col0 = blockIdx.y * 128 + w * 16;       // 8 waves cover 128 cols
  int b = blockIdx.z;

  v8f accf = {};
  const _Float16* frow = fuseF + (size_t)rt * 16 * 2048;
  const _Float16* vrow = vp + (size_t)(b * MPT + col0) * 2048;
  for (int kc = 0; kc < 64; ++kc) {
    v16h a  = load_frag_a(frow + kc * 32, 2048);
    v16h bm = load_frag_b(vrow + kc * 32, 2048);
    accf = __builtin_amdgcn_wmma_f32_16x16x32_f16(false, a, false, bm, (short)0, accf, false, false);
  }
  v8f accg = {};
  const _Float16* grow = gateF + (size_t)rt * 16 * 512;
  const _Float16* trow = featT + (size_t)(b * MPT + col0) * 512;
  for (int kc = 0; kc < 16; ++kc) {
    v16h a  = load_frag_a(grow + kc * 32, 512);
    v16h bm = load_frag_b(trow + kc * 32, 512);
    accg = __builtin_amdgcn_wmma_f32_16x16x32_f16(false, a, false, bm, (short)0, accg, false, false);
  }
  int n = col0 + (lane & 15);
  int mh = 8 * (lane >> 4);
  int fidx = fps_idx[b * MPT + n];
#pragma unroll
  for (int r = 0; r < 8; ++r) {
    int M = rt * 16 + r + mh;
    float fg = seed_feat[((size_t)b * CSEED + M) * NPTS + fidx];   // exact f32 feat_g
    float gt = 1.f / (1.f + __expf(-(accg[r] + gate_b[M])));
    out[((size_t)b * CSEED + M) * MPT + n] = accf[r] + fuse_b[M] + gt * fg;
  }
}

extern "C" void kernel_launch(void* const* d_in, const int* in_sizes, int n_in,
                              void* d_out, int out_size, void* d_ws, size_t ws_size,
                              hipStream_t stream) {
  (void)in_sizes; (void)n_in; (void)out_size; (void)ws_size;
  const float* seed_xyz  = (const float*)d_in[0];
  const float* seed_feat = (const float*)d_in[1];
  const int*   mask      = (const int*)d_in[2];
  const float* views_rot = (const float*)d_in[3];
  const float* crop_w1   = (const float*)d_in[4];
  const float* crop_b1   = (const float*)d_in[5];
  const float* crop_w2   = (const float*)d_in[6];
  const float* crop_b2   = (const float*)d_in[7];
  const float* fuse_w    = (const float*)d_in[8];
  const float* fuse_b    = (const float*)d_in[9];
  const float* gate_w    = (const float*)d_in[10];
  const float* gate_b    = (const float*)d_in[11];
  float* out = (float*)d_out;

  char* ws = (char*)d_ws;
  size_t off = 0;
  auto alloc = [&](size_t bytes) -> void* {
    void* p = ws + off;
    off = (off + bytes + 255) & ~(size_t)255;
    return p;
  };
  int*      fps_idx = (int*)alloc((size_t)B * MPT * 4);
  float*    xyz_g   = (float*)alloc((size_t)B * MPT * 3 * 4);
  _Float16* featT   = (_Float16*)alloc((size_t)B * MPT * CSEED * 2);
  _Float16* vp      = (_Float16*)alloc((size_t)B * MPT * 2048 * 2);
  int*      grp     = (int*)alloc((size_t)B * MPT * 144 * 4);
  _Float16* gxyz    = (_Float16*)alloc((size_t)B * MPT * 144 * 3 * 2);
  _Float16* W1f     = (_Float16*)alloc((size_t)4 * 256 * K1PAD * 2);
  _Float16* W2f     = (_Float16*)alloc((size_t)4 * 512 * 256 * 2);
  _Float16* fuseF   = (_Float16*)alloc((size_t)512 * 2048 * 2);
  _Float16* gateF   = (_Float16*)alloc((size_t)512 * 512 * 2);

  { int n = 4 * 256 * K1PAD; convert_w1_kernel<<<(n + 255) / 256, 256, 0, stream>>>(crop_w1, W1f); }
  { int n = 4 * 512 * 256;   convert_f16_kernel<<<(n + 255) / 256, 256, 0, stream>>>(crop_w2, W2f, n); }
  { int n = 512 * 2048;      convert_f16_kernel<<<(n + 255) / 256, 256, 0, stream>>>(fuse_w, fuseF, n); }
  { int n = 512 * 512;       convert_f16_kernel<<<(n + 255) / 256, 256, 0, stream>>>(gate_w, gateF, n); }

  fps_kernel<<<B, 1024, 0, stream>>>(seed_xyz, mask, fps_idx);
  gather_kernel<<<dim3(MPT, B), CSEED, 0, stream>>>(seed_xyz, seed_feat, fps_idx, xyz_g, featT);
  local_group_kernel<<<dim3(MPT, B), 1024, 0, stream>>>(xyz_g, views_rot, grp, gxyz);

  const int roff[4] = {0, 2048 * 16, 2048 * 48, 2048 * 80};  // sample-slot offsets
  crop_mlp_kernel<16><<<B * MPT, 256, 0, stream>>>(
      W1f + (size_t)0 * 256 * K1PAD, crop_b1 + 0 * 256,
      W2f + (size_t)0 * 512 * 256,   crop_b2 + 0 * 512,
      featT, gxyz + (size_t)roff[0] * 3, grp + roff[0], vp, 0 * 512);
  crop_mlp_kernel<32><<<B * MPT, 256, 0, stream>>>(
      W1f + (size_t)1 * 256 * K1PAD, crop_b1 + 1 * 256,
      W2f + (size_t)1 * 512 * 256,   crop_b2 + 1 * 512,
      featT, gxyz + (size_t)roff[1] * 3, grp + roff[1], vp, 1 * 512);
  crop_mlp_kernel<32><<<B * MPT, 256, 0, stream>>>(
      W1f + (size_t)2 * 256 * K1PAD, crop_b1 + 2 * 256,
      W2f + (size_t)2 * 512 * 256,   crop_b2 + 2 * 512,
      featT, gxyz + (size_t)roff[2] * 3, grp + roff[2], vp, 2 * 512);
  crop_mlp_kernel<64><<<B * MPT, 256, 0, stream>>>(
      W1f + (size_t)3 * 256 * K1PAD, crop_b1 + 3 * 256,
      W2f + (size_t)3 * 512 * 256,   crop_b2 + 3 * 512,
      featT, gxyz + (size_t)roff[3] * 3, grp + roff[3], vp, 3 * 512);

  fuse_gate_kernel<<<dim3(32, 8, B), 256, 0, stream>>>(
      fuseF, fuse_b, gateF, gate_b, vp, featT, seed_feat, fps_idx, out);
}
